// gat_24756191494287
// MI455X (gfx1250) — compile-verified, matrix operand order checked by reading, not run on hardware
//
#include <hip/hip_runtime.h>
#include <hip/hip_bf16.h>

#define N_NODES 50000
#define N_EDGES 800000
#define IN_DIM  128
#define HEADS   8
#define F_OUT   16
#define HF      128   // HEADS*F_OUT

typedef _Float16 v16h __attribute__((ext_vector_type(16)));
typedef _Float16 v8h  __attribute__((ext_vector_type(8)));
typedef float    v8f  __attribute__((ext_vector_type(8)));

// ---- monotonic float<->uint encoding for atomicMax-based segment max ----
__device__ __forceinline__ unsigned fenc(float x) {
    unsigned u = __float_as_uint(x);
    return (u & 0x80000000u) ? ~u : (u | 0x80000000u);
}
__device__ __forceinline__ float fdec(unsigned k) {
    unsigned u = (k & 0x80000000u) ? (k & 0x7FFFFFFFu) : ~k;
    return __uint_as_float(u);
}
__device__ __forceinline__ float lrelu(float x) { return x > 0.f ? x : 0.2f * x; }

// ---------------------------------------------------------------------------
// 0) convert both weight matrices fp32 -> fp16 (done every call; tiny)
// ---------------------------------------------------------------------------
__global__ __launch_bounds__(256) void cvt_weights_kernel(
    const float* __restrict__ Wp, const float* __restrict__ Wsk,
    _Float16* __restrict__ Wp16, _Float16* __restrict__ Ws16) {
    int i = blockIdx.x * 256 + threadIdx.x;
    if (i < IN_DIM * HF) {
        Wp16[i] = (_Float16)Wp[i];
        Ws16[i] = (_Float16)Wsk[i];
    }
}

// ---------------------------------------------------------------------------
// 1) init segment-max keys and denominators
// ---------------------------------------------------------------------------
__global__ __launch_bounds__(256) void init_kernel(unsigned* __restrict__ emax,
                                                   float* __restrict__ denom) {
    int i = blockIdx.x * 256 + threadIdx.x;
    if (i < N_NODES * HEADS) {
        emax[i]  = 0u;     // lowest key (decodes below any real value's key)
        denom[i] = 0.f;
    }
}

// ---------------------------------------------------------------------------
// 2) dual GEMM via WMMA:  proj = X @ Wp^T ;  accum = X @ Ws^T + bias
//    block = 256 threads = 8 waves; block handles a 16-row node tile,
//    wave w handles output columns [16w, 16w+16) == head w. K=128 in 4 steps.
// ---------------------------------------------------------------------------
__global__ __launch_bounds__(256) void gemm_kernel(
    const float*    __restrict__ x,
    const _Float16* __restrict__ Wp16,
    const _Float16* __restrict__ Ws16,
    const float*    __restrict__ bias,
    float*          __restrict__ proj,
    float*          __restrict__ accum) {
    __shared__ __align__(16) _Float16 ldsX[16 * 136];   // padded rows: bank-spread

    const int tid     = threadIdx.x;
    const int rowbase = blockIdx.x * 16;

    // stage 16x128 fp32 tile of X into LDS as fp16 (8 elems / thread)
    #pragma unroll
    for (int i = 0; i < 8; ++i) {
        int idx = tid + i * 256;
        int r = idx >> 7, c = idx & 127;
        ldsX[r * 136 + c] = (_Float16)x[(rowbase + r) * IN_DIM + c];
    }
    __syncthreads();

    const int wave = tid >> 5;          // 0..7  => head / column tile
    const int lane = tid & 31;
    const int nloc = lane & 15;         // column within tile (and A-row)
    const int kb   = (lane >> 4) << 3;  // K sub-offset: 0 (lanes 0-15) / 8 (16-31)
    const int ncol = wave * 16 + nloc;  // global output column == row of W

    v8f cP = {};
    v8f cS = {};
    #pragma unroll
    for (int c = 0; c < 4; ++c) {       // K chunks of 32
        const int kc = c * 32 + kb;
        // A fragment from LDS: row nloc, K = kc+0..7 and kc+16..23
        const _Float16* pa = &ldsX[nloc * 136 + kc];
        v8h alo = *(const v8h*)(pa);
        v8h ahi = *(const v8h*)(pa + 16);
        v16h a = __builtin_shufflevector(alo, ahi, 0,1,2,3,4,5,6,7,8,9,10,11,12,13,14,15);
        // B fragments from fp16 weights: row ncol of W (== column of W^T)
        const _Float16* pb = &Wp16[ncol * IN_DIM + kc];
        v8h blo = *(const v8h*)(pb);
        v8h bhi = *(const v8h*)(pb + 16);
        v16h b = __builtin_shufflevector(blo, bhi, 0,1,2,3,4,5,6,7,8,9,10,11,12,13,14,15);
        cP = __builtin_amdgcn_wmma_f32_16x16x32_f16(false, a, false, b, (short)0, cP, false, false);

        const _Float16* ps = &Ws16[ncol * IN_DIM + kc];
        v8h slo = *(const v8h*)(ps);
        v8h shi = *(const v8h*)(ps + 16);
        v16h bs = __builtin_shufflevector(slo, shi, 0,1,2,3,4,5,6,7,8,9,10,11,12,13,14,15);
        cS = __builtin_amdgcn_wmma_f32_16x16x32_f16(false, a, false, bs, (short)0, cS, false, false);
    }

    const float bv   = bias[ncol];
    const int   rtop = rowbase + ((lane >> 4) << 3);  // D rows: v (+8 upper lanes)
    #pragma unroll
    for (int v = 0; v < 8; ++v) {
        int row = rtop + v;
        proj [row * HF + ncol] = cP[v];
        accum[row * HF + ncol] = cS[v] + bv;          // skip-proj + bias seed
    }
}

// ---------------------------------------------------------------------------
// 3) per-(node,head) attention partial scores
// ---------------------------------------------------------------------------
__global__ __launch_bounds__(256) void scores_kernel(
    const float* __restrict__ proj, const float* __restrict__ a_src,
    const float* __restrict__ a_tgt, float* __restrict__ ssrc,
    float* __restrict__ stgt) {
    int t = blockIdx.x * 256 + threadIdx.x;
    if (t >= N_NODES * HEADS) return;
    int h = t & 7, n = t >> 3;
    const float4* p  = (const float4*)&proj [n * HF + h * F_OUT];
    const float4* as = (const float4*)&a_src[h * F_OUT];
    const float4* at = (const float4*)&a_tgt[h * F_OUT];
    float s1 = 0.f, s2 = 0.f;
    #pragma unroll
    for (int i = 0; i < 4; ++i) {
        float4 v = p[i], a = as[i], b = at[i];
        s1 += v.x * a.x + v.y * a.y + v.z * a.z + v.w * a.w;
        s2 += v.x * b.x + v.y * b.y + v.z * b.z + v.w * b.w;
    }
    ssrc[t] = s1;
    stgt[t] = s2;
}

// ---------------------------------------------------------------------------
// 4) segment max over incoming edges (L2-resident atomicMax on monotonic keys)
// ---------------------------------------------------------------------------
__global__ __launch_bounds__(256) void edge_max_kernel(
    const int* __restrict__ ei, const float* __restrict__ ssrc,
    const float* __restrict__ stgt, unsigned* __restrict__ emax) {
    int e = blockIdx.x * 256 + threadIdx.x;
    if (e >= N_EDGES) return;
    int s = ei[e], t = ei[N_EDGES + e];
    const float4* ps = (const float4*)&ssrc[s * HEADS];
    const float4* pt = (const float4*)&stgt[t * HEADS];
    float4 a0 = ps[0], a1 = ps[1], b0 = pt[0], b1 = pt[1];
    float eh[8] = {a0.x + b0.x, a0.y + b0.y, a0.z + b0.z, a0.w + b0.w,
                   a1.x + b1.x, a1.y + b1.y, a1.z + b1.z, a1.w + b1.w};
    #pragma unroll
    for (int h = 0; h < 8; ++h)
        atomicMax(&emax[t * HEADS + h], fenc(lrelu(eh[h])));
}

// ---------------------------------------------------------------------------
// 5) segment sum of exp(e - max)
// ---------------------------------------------------------------------------
__global__ __launch_bounds__(256) void edge_sum_kernel(
    const int* __restrict__ ei, const float* __restrict__ ssrc,
    const float* __restrict__ stgt, const unsigned* __restrict__ emax,
    float* __restrict__ denom) {
    int e = blockIdx.x * 256 + threadIdx.x;
    if (e >= N_EDGES) return;
    int s = ei[e], t = ei[N_EDGES + e];
    const float4* ps = (const float4*)&ssrc[s * HEADS];
    const float4* pt = (const float4*)&stgt[t * HEADS];
    float4 a0 = ps[0], a1 = ps[1], b0 = pt[0], b1 = pt[1];
    float eh[8] = {a0.x + b0.x, a0.y + b0.y, a0.z + b0.z, a0.w + b0.w,
                   a1.x + b1.x, a1.y + b1.y, a1.z + b1.z, a1.w + b1.w};
    #pragma unroll
    for (int h = 0; h < 8; ++h) {
        float m = fdec(emax[t * HEADS + h]);
        atomicAdd(&denom[t * HEADS + h], __expf(lrelu(eh[h]) - m));
    }
}

// ---------------------------------------------------------------------------
// 6) weighted scatter-add of projected source features (one thread = edge,head)
// ---------------------------------------------------------------------------
__global__ __launch_bounds__(256) void scatter_kernel(
    const int* __restrict__ ei, const float* __restrict__ ssrc,
    const float* __restrict__ stgt, const unsigned* __restrict__ emax,
    const float* __restrict__ denom, const float* __restrict__ proj,
    float* __restrict__ accum) {
    int tt = blockIdx.x * 256 + threadIdx.x;
    if (tt >= N_EDGES * HEADS) return;
    int e = tt >> 3, h = tt & 7;
    int s = ei[e], t = ei[N_EDGES + e];
    float v = lrelu(ssrc[s * HEADS + h] + stgt[t * HEADS + h]);
    float m = fdec(emax[t * HEADS + h]);
    float alpha = __expf(v - m) / (denom[t * HEADS + h] + 1e-16f);
    const float4* p = (const float4*)&proj[s * HF + h * F_OUT];
    float* o = &accum[t * HF + h * F_OUT];
    #pragma unroll
    for (int i = 0; i < 4; ++i) {
        float4 q = p[i];
        atomicAdd(o + 4 * i + 0, q.x * alpha);
        atomicAdd(o + 4 * i + 1, q.y * alpha);
        atomicAdd(o + 4 * i + 2, q.z * alpha);
        atomicAdd(o + 4 * i + 3, q.w * alpha);
    }
}

// ---------------------------------------------------------------------------
// 7) ELU in place on d_out
// ---------------------------------------------------------------------------
__global__ __launch_bounds__(256) void elu_kernel(float* __restrict__ out) {
    int i = blockIdx.x * 256 + threadIdx.x;
    if (i < N_NODES * HF) {
        float x = out[i];
        out[i] = x > 0.f ? x : (__expf(x) - 1.f);
    }
}

// ---------------------------------------------------------------------------
extern "C" void kernel_launch(void* const* d_in, const int* in_sizes, int n_in,
                              void* d_out, int out_size, void* d_ws, size_t ws_size,
                              hipStream_t stream) {
    const float* x      = (const float*)d_in[0];
    const int*   ei     = (const int*)  d_in[1];
    const float* W_proj = (const float*)d_in[2];
    const float* a_src  = (const float*)d_in[3];
    const float* a_tgt  = (const float*)d_in[4];
    const float* W_skip = (const float*)d_in[5];
    const float* bias   = (const float*)d_in[6];
    float* out = (float*)d_out;   // doubles as the accumulator

    // carve workspace (256B-aligned slices)
    auto alignup = [](size_t v) { return (v + 255) & ~size_t(255); };
    char* ws = (char*)d_ws;
    size_t o = 0;
    _Float16* Wp16 = (_Float16*)(ws + o); o += alignup(IN_DIM * HF * sizeof(_Float16));
    _Float16* Ws16 = (_Float16*)(ws + o); o += alignup(IN_DIM * HF * sizeof(_Float16));
    float*    proj = (float*)   (ws + o); o += alignup((size_t)N_NODES * HF * sizeof(float));
    float*    ssrc = (float*)   (ws + o); o += alignup((size_t)N_NODES * HEADS * sizeof(float));
    float*    stgt = (float*)   (ws + o); o += alignup((size_t)N_NODES * HEADS * sizeof(float));
    unsigned* emax = (unsigned*)(ws + o); o += alignup((size_t)N_NODES * HEADS * sizeof(unsigned));
    float*    denm = (float*)   (ws + o); o += alignup((size_t)N_NODES * HEADS * sizeof(float));
    (void)o; (void)ws_size; (void)in_sizes; (void)n_in; (void)out_size;

    cvt_weights_kernel<<<(IN_DIM * HF + 255) / 256, 256, 0, stream>>>(W_proj, W_skip, Wp16, Ws16);
    init_kernel<<<(N_NODES * HEADS + 255) / 256, 256, 0, stream>>>(emax, denm);
    gemm_kernel<<<N_NODES / 16, 256, 0, stream>>>(x, Wp16, Ws16, bias, proj, out);
    scores_kernel<<<(N_NODES * HEADS + 255) / 256, 256, 0, stream>>>(proj, a_src, a_tgt, ssrc, stgt);
    edge_max_kernel<<<(N_EDGES + 255) / 256, 256, 0, stream>>>(ei, ssrc, stgt, emax);
    edge_sum_kernel<<<(N_EDGES + 255) / 256, 256, 0, stream>>>(ei, ssrc, stgt, emax, denm);
    scatter_kernel<<<(N_EDGES * HEADS + 255) / 256, 256, 0, stream>>>(ei, ssrc, stgt, emax, denm, proj, out);
    elu_kernel<<<(N_NODES * HF + 255) / 256, 256, 0, stream>>>(out);
}
